// CausalSelfAttention_5643587027299
// MI455X (gfx1250) — matrix operand order repaired and checked by reading, not compiled, not run on hardware
//
#include <hip/hip_runtime.h>

// ---------------------------------------------------------------------------
// CDNA5 (gfx1250, wave32) causal self-attention forward.
// All matmuls on v_wmma_f32_16x16x32_bf16 (f32 accumulate).
// Async global->LDS staging (ASYNCcnt) where tiles are raw bf16 copies.
// ---------------------------------------------------------------------------

typedef __attribute__((ext_vector_type(16))) __bf16 v16bf;
typedef __attribute__((ext_vector_type(8)))  float  v8f;

#define BB 2
#define TT 2048
#define CC 768
#define HH 12
#define DD 64
#define N3C (3 * CC)
#define MM (BB * TT)          // 4096 rows

#define WMMA_BF16(a, b, c) \
    __builtin_amdgcn_wmma_f32_16x16x32_bf16(false, (a), false, (b), (short)0, (c), false, false)

// async copy of 16 bytes global -> LDS (tracked by ASYNCcnt)
__device__ inline void async_cp16(void* lds_ptr, const void* gptr) {
    const unsigned loff = (unsigned)(unsigned long long)lds_ptr;   // low 32 bits = LDS offset
    const unsigned long long g = (unsigned long long)gptr;
    asm volatile("global_load_async_to_lds_b128 %0, %1, off" :: "v"(loff), "v"(g) : "memory");
}
__device__ inline void async_wait0() {
    asm volatile("s_wait_asynccnt 0" ::: "memory");
}

// ---- fragment loaders (wave32 layouts per CDNA5 ISA 7.12.2) ----------------
// A (16x32 bf16, row-major source, row stride ld elements):
//   lane L: m = L%16; half = L/16; VGPR h holds K pair  k0 = (h%4)*2 + (h/4)*16 + half*8
__device__ inline v16bf ld_a_frag(const __bf16* base, int ld, int lane) {
    const int m  = lane & 15;
    const int kb = (lane >> 4) * 8;
    const __bf16* row = base + m * ld;
    v16bf a;
#pragma unroll
    for (int h = 0; h < 8; ++h) {
        const int k0 = ((h & 3) * 2) + ((h >> 2) * 16) + kb;
        a[2 * h + 0] = row[k0];
        a[2 * h + 1] = row[k0 + 1];
    }
    return a;
}

// B (32x16 bf16): lane L: n = L%16; K = (L/16)*8 + (i%8) + (i/8)*16
__device__ inline v16bf ld_b_frag(const __bf16* base, int ld, int lane) {
    const int n  = lane & 15;
    const int kb = (lane >> 4) * 8;
    const __bf16* row = base + n * ld + kb;
    v16bf b;
#pragma unroll
    for (int i = 0; i < 16; ++i) b[i] = row[(i & 7) + ((i >> 3) << 4)];
    return b;
}

// ---------------------------------------------------------------------------
// Kernel 1: fused QKV projection.  qkv[m,f] = x[m,:] . Wqkv[f,:] + bqkv[f]
// Block tile 128x64, 4 waves; each wave: 32x64 C tile (2 A-frags, B reused).
// Grid: (N3C/64 = 36, MM/128 = 32), 128 threads.
// ---------------------------------------------------------------------------
__global__ __launch_bounds__(128)
void qkv_gemm_kernel(const float* __restrict__ x, const float* __restrict__ Wqkv,
                     const float* __restrict__ bqkv,
                     unsigned short* __restrict__ qws, unsigned short* __restrict__ kws,
                     unsigned short* __restrict__ vws) {
    __shared__ __attribute__((aligned(16))) __bf16 As[128][40];
    __shared__ __attribute__((aligned(16))) __bf16 Bs[64][40];

    const int tid  = threadIdx.x;
    const int lane = tid & 31;
    const int wv   = tid >> 5;
    const int n0   = blockIdx.x * 64;
    const int m0   = blockIdx.y * 128;

    v8f acc[2][4];
#pragma unroll
    for (int mi = 0; mi < 2; ++mi)
#pragma unroll
        for (int nt = 0; nt < 4; ++nt)
#pragma unroll
            for (int r = 0; r < 8; ++r) acc[mi][nt][r] = 0.0f;

    for (int kc = 0; kc < CC; kc += 32) {
        // stage A (x, 128x32) and B (Wqkv, 64x32), f32 -> bf16
        for (int idx = tid; idx < 128 * 8; idx += 128) {
            const int r  = idx >> 3;
            const int c4 = (idx & 7) * 4;
            const float4 xa = *(const float4*)(x + (size_t)(m0 + r) * CC + kc + c4);
            As[r][c4 + 0] = (__bf16)xa.x; As[r][c4 + 1] = (__bf16)xa.y;
            As[r][c4 + 2] = (__bf16)xa.z; As[r][c4 + 3] = (__bf16)xa.w;
        }
        for (int idx = tid; idx < 64 * 8; idx += 128) {
            const int r  = idx >> 3;
            const int c4 = (idx & 7) * 4;
            const float4 wa = *(const float4*)(Wqkv + (size_t)(n0 + r) * CC + kc + c4);
            Bs[r][c4 + 0] = (__bf16)wa.x; Bs[r][c4 + 1] = (__bf16)wa.y;
            Bs[r][c4 + 2] = (__bf16)wa.z; Bs[r][c4 + 3] = (__bf16)wa.w;
        }
        __syncthreads();

        const v16bf a0 = ld_a_frag(&As[32 * wv + 0][0],  40, lane);
        const v16bf a1 = ld_a_frag(&As[32 * wv + 16][0], 40, lane);
#pragma unroll
        for (int nt = 0; nt < 4; ++nt) {
            const v16bf bfrag = ld_b_frag(&Bs[nt * 16][0], 40, lane);
            acc[0][nt] = WMMA_BF16(a0, bfrag, acc[0][nt]);
            acc[1][nt] = WMMA_BF16(a1, bfrag, acc[1][nt]);
        }
        __syncthreads();
    }

    // epilogue: + bias, scatter into q/k/v [B,H,T,D] (bf16)
    const int hf  = lane >> 4;
    const int col = lane & 15;
#pragma unroll
    for (int nt = 0; nt < 4; ++nt) {
        const int f = n0 + nt * 16 + col;
        const float bias = bqkv[f];
        const int which = f / CC;
        const int c = f % CC;
        const int h = c >> 6;      // / DD
        const int d = c & 63;      // % DD
        __bf16* dst = (__bf16*)(which == 0 ? qws : (which == 1 ? kws : vws));
#pragma unroll
        for (int mi = 0; mi < 2; ++mi)
#pragma unroll
            for (int r = 0; r < 8; ++r) {
                const int m = m0 + 32 * wv + 16 * mi + r + 8 * hf;
                const int bb = m >> 11;        // / TT
                const int t  = m & (TT - 1);
                dst[(((size_t)(bb * HH + h)) * TT + t) * DD + d] =
                    (__bf16)(acc[mi][nt][r] + bias);
            }
    }
}

// ---------------------------------------------------------------------------
// Kernel 2: flash attention, 64 query rows per block, 4 waves (16 rows each).
// K tile staged via async global->LDS; V staged transposed via VGPR scatter.
// Grid: (TT/64 = 32, BB*HH = 24), 128 threads.
// ---------------------------------------------------------------------------
__global__ __launch_bounds__(128)
void attn_kernel(const unsigned short* __restrict__ qws, const unsigned short* __restrict__ kws,
                 const unsigned short* __restrict__ vws, unsigned short* __restrict__ yws,
                 const int* __restrict__ is_causal) {
    __shared__ __attribute__((aligned(16))) __bf16 Kt[64][72];        // [time][d]
    __shared__ __attribute__((aligned(16))) __bf16 Vt[64][72];        // transposed: [d][time]
    __shared__ __attribute__((aligned(16))) __bf16 Pt[4][16][72];     // per-wave P staging

    const int tid  = threadIdx.x;
    const int lane = tid & 31;
    const int wv   = tid >> 5;
    const int bh   = blockIdx.y;
    const int q0   = blockIdx.x * 64;
    const int causal = *is_causal;

    const __bf16* Q = (const __bf16*)qws + (size_t)bh * TT * DD;
    const __bf16* K = (const __bf16*)kws + (size_t)bh * TT * DD;
    const __bf16* V = (const __bf16*)vws + (size_t)bh * TT * DD;

    // Q fragments for this wave's 16 rows: resident for the whole loop
    const v16bf aq0 = ld_a_frag(Q + (size_t)(q0 + 16 * wv) * DD + 0,  DD, lane);
    const v16bf aq1 = ld_a_frag(Q + (size_t)(q0 + 16 * wv) * DD + 32, DD, lane);

    v8f o[4];
    float mrow[8], lrow[8];
#pragma unroll
    for (int dt = 0; dt < 4; ++dt)
#pragma unroll
        for (int r = 0; r < 8; ++r) o[dt][r] = 0.0f;
#pragma unroll
    for (int r = 0; r < 8; ++r) { mrow[r] = -3.0e38f; lrow[r] = 0.0f; }

    const int hf  = lane >> 4;
    const int col = lane & 15;
    const int kend = causal ? (q0 + 63) : (TT - 1);
    const float scale = 0.125f;   // 1/sqrt(64)

    for (int k0 = 0; k0 <= kend; k0 += 64) {
        // ---- K tile: async global->LDS (row-major bf16 copy, ASYNCcnt) ----
        for (int idx = tid; idx < 64 * 8; idx += 128) {
            const int r  = idx >> 3;
            const int c8 = (idx & 7) * 8;
            async_cp16(&Kt[r][c8], K + (size_t)(k0 + r) * DD + c8);
        }
        // ---- V tile: load + transpose scatter into LDS ----
        for (int idx = tid; idx < 64 * 8; idx += 128) {
            const int r  = idx >> 3;
            const int c8 = (idx & 7) * 8;
            const uint4 vv = *(const uint4*)(V + (size_t)(k0 + r) * DD + c8);
            const __bf16* vb = (const __bf16*)&vv;
#pragma unroll
            for (int j = 0; j < 8; ++j) Vt[c8 + j][r] = vb[j];
        }
        async_wait0();
        __syncthreads();

        // ---- S = Q . K^T  (16x64 per wave) ----
        v8f sc[4];
#pragma unroll
        for (int nt = 0; nt < 4; ++nt) {
#pragma unroll
            for (int r = 0; r < 8; ++r) sc[nt][r] = 0.0f;
            const v16bf bk0 = ld_b_frag(&Kt[nt * 16][0],  72, lane);
            sc[nt] = WMMA_BF16(aq0, bk0, sc[nt]);
            const v16bf bk1 = ld_b_frag(&Kt[nt * 16][32], 72, lane);
            sc[nt] = WMMA_BF16(aq1, bk1, sc[nt]);
        }

        // ---- scale + causal mask ----
#pragma unroll
        for (int nt = 0; nt < 4; ++nt)
#pragma unroll
            for (int r = 0; r < 8; ++r) {
                float s = sc[nt][r] * scale;
                if (causal) {
                    const int qrow = q0 + 16 * wv + r + 8 * hf;
                    const int kcol = k0 + nt * 16 + col;
                    if (kcol > qrow) s = -3.0e38f;
                }
                sc[nt][r] = s;
            }

        // ---- online softmax (per-row over 16-lane half + 4 n-tiles) ----
#pragma unroll
        for (int r = 0; r < 8; ++r) {
            float mx = fmaxf(fmaxf(sc[0][r], sc[1][r]), fmaxf(sc[2][r], sc[3][r]));
#pragma unroll
            for (int off = 1; off < 16; off <<= 1) mx = fmaxf(mx, __shfl_xor(mx, off, 32));
            const float mnew  = fmaxf(mrow[r], mx);
            const float alpha = __expf(mrow[r] - mnew);
            mrow[r] = mnew;
            float rs = 0.0f;
#pragma unroll
            for (int nt = 0; nt < 4; ++nt) {
                const float p = __expf(sc[nt][r] - mnew);
                sc[nt][r] = p;
                rs += p;
            }
#pragma unroll
            for (int off = 1; off < 16; off <<= 1) rs += __shfl_xor(rs, off, 32);
            lrow[r] = lrow[r] * alpha + rs;
#pragma unroll
            for (int dt = 0; dt < 4; ++dt) o[dt][r] *= alpha;
        }

        // ---- re-layout P (D-layout -> A-fragment) through per-wave LDS ----
#pragma unroll
        for (int nt = 0; nt < 4; ++nt)
#pragma unroll
            for (int r = 0; r < 8; ++r)
                Pt[wv][r + 8 * hf][nt * 16 + col] = (__bf16)sc[nt][r];
        asm volatile("s_wait_dscnt 0" ::: "memory");   // wave-local LDS W->R ordering

        const v16bf ap0 = ld_a_frag(&Pt[wv][0][0],  72, lane);
        const v16bf ap1 = ld_a_frag(&Pt[wv][0][32], 72, lane);

        // ---- O += P . V ----
#pragma unroll
        for (int dt = 0; dt < 4; ++dt) {
            const v16bf bv0 = ld_b_frag(&Vt[dt * 16][0],  72, lane);
            o[dt] = WMMA_BF16(ap0, bv0, o[dt]);
            const v16bf bv1 = ld_b_frag(&Vt[dt * 16][32], 72, lane);
            o[dt] = WMMA_BF16(ap1, bv1, o[dt]);
        }
        __syncthreads();
    }

    // ---- epilogue: y[b, t, h*64+d] = O/l  (bf16) ----
    const int bidx = bh / HH;
    const int h    = bh % HH;
    __bf16* Y = (__bf16*)yws;
#pragma unroll
    for (int dt = 0; dt < 4; ++dt)
#pragma unroll
        for (int r = 0; r < 8; ++r) {
            const int t = q0 + 16 * wv + r + 8 * hf;
            const float val = o[dt][r] / lrow[r];
            Y[((size_t)bidx * TT + t) * CC + h * DD + dt * 16 + col] = (__bf16)val;
        }
}

// ---------------------------------------------------------------------------
// Kernel 3: output projection.  out[m,f] = y[m,:] . Wproj[f,:] + bproj[f]
// Block tile 128x64, 4 waves; A (y, bf16) staged via async global->LDS.
// Grid: (CC/64 = 12, MM/128 = 32), 128 threads.
// ---------------------------------------------------------------------------
__global__ __launch_bounds__(128)
void proj_gemm_kernel(const unsigned short* __restrict__ yws, const float* __restrict__ Wp,
                      const float* __restrict__ bp, float* __restrict__ out) {
    __shared__ __attribute__((aligned(16))) __bf16 As[128][40];
    __shared__ __attribute__((aligned(16))) __bf16 Bs[64][40];

    const int tid  = threadIdx.x;
    const int lane = tid & 31;
    const int wv   = tid >> 5;
    const int n0   = blockIdx.x * 64;
    const int m0   = blockIdx.y * 128;
    const __bf16* Y = (const __bf16*)yws;

    v8f acc[2][4];
#pragma unroll
    for (int mi = 0; mi < 2; ++mi)
#pragma unroll
        for (int nt = 0; nt < 4; ++nt)
#pragma unroll
            for (int r = 0; r < 8; ++r) acc[mi][nt][r] = 0.0f;

    for (int kc = 0; kc < CC; kc += 32) {
        // A tile (bf16, 128x32): async raw copy global->LDS
        for (int idx = tid; idx < 128 * 4; idx += 128) {
            const int r  = idx >> 2;
            const int c8 = (idx & 3) * 8;
            async_cp16(&As[r][c8], Y + (size_t)(m0 + r) * CC + kc + c8);
        }
        // B tile (Wproj, 64x32): f32 -> bf16
        for (int idx = tid; idx < 64 * 8; idx += 128) {
            const int r  = idx >> 3;
            const int c4 = (idx & 7) * 4;
            const float4 wa = *(const float4*)(Wp + (size_t)(n0 + r) * CC + kc + c4);
            Bs[r][c4 + 0] = (__bf16)wa.x; Bs[r][c4 + 1] = (__bf16)wa.y;
            Bs[r][c4 + 2] = (__bf16)wa.z; Bs[r][c4 + 3] = (__bf16)wa.w;
        }
        async_wait0();
        __syncthreads();

        const v16bf a0 = ld_a_frag(&As[32 * wv + 0][0],  40, lane);
        const v16bf a1 = ld_a_frag(&As[32 * wv + 16][0], 40, lane);
#pragma unroll
        for (int nt = 0; nt < 4; ++nt) {
            const v16bf bfrag = ld_b_frag(&Bs[nt * 16][0], 40, lane);
            acc[0][nt] = WMMA_BF16(a0, bfrag, acc[0][nt]);
            acc[1][nt] = WMMA_BF16(a1, bfrag, acc[1][nt]);
        }
        __syncthreads();
    }

    const int hf  = lane >> 4;
    const int col = lane & 15;
#pragma unroll
    for (int nt = 0; nt < 4; ++nt) {
        const int f = n0 + nt * 16 + col;
        const float bias = bp[f];
#pragma unroll
        for (int mi = 0; mi < 2; ++mi)
#pragma unroll
            for (int r = 0; r < 8; ++r) {
                const int m = m0 + 32 * wv + 16 * mi + r + 8 * hf;
                out[(size_t)m * CC + f] = acc[mi][nt][r] + bias;
            }
    }
}

// ---------------------------------------------------------------------------
extern "C" void kernel_launch(void* const* d_in, const int* in_sizes, int n_in,
                              void* d_out, int out_size, void* d_ws, size_t ws_size,
                              hipStream_t stream) {
    const float* x     = (const float*)d_in[0];
    const float* Wqkv  = (const float*)d_in[1];
    const float* bqkv  = (const float*)d_in[2];
    const float* Wproj = (const float*)d_in[3];
    const float* bproj = (const float*)d_in[4];
    const int* is_causal = (const int*)d_in[5];
    float* out = (float*)d_out;

    const size_t per = (size_t)BB * HH * TT * DD;   // 3,145,728 bf16 elements each
    unsigned short* qw = (unsigned short*)d_ws;
    unsigned short* kw = qw + per;
    unsigned short* vw = kw + per;
    unsigned short* yw = vw + per;                  // total 4*per*2 = ~25.2 MB

    qkv_gemm_kernel<<<dim3(N3C / 64, MM / 128), 128, 0, stream>>>(x, Wqkv, bqkv, qw, kw, vw);
    attn_kernel<<<dim3(TT / 64, BB * HH), 128, 0, stream>>>(qw, kw, vw, yw, is_causal);
    proj_gemm_kernel<<<dim3(CC / 64, MM / 128), 128, 0, stream>>>(yw, Wproj, bproj, out);
}
